// PseudoGroupContrast_20186346291803
// MI455X (gfx1250) — matrix operand order repaired and checked by reading, not compiled, not run on hardware
//
#include <hip/hip_runtime.h>
#include <hip/hip_bf16.h>
#include <stdint.h>

typedef __attribute__((ext_vector_type(16))) __bf16 v16bf;
typedef __attribute__((ext_vector_type(8)))  float  v8f;

#define PROJ 128
#define QS   150
#define CLS  7
#define NQ   (QS*CLS)        /* 1050 */
#define NCT  66              /* ceil(1050/16) column tiles (padded to 1056) */
#define TILE_BYTES 4096      /* one 16-col tile: 128(K) x 16(N) bf16 */
#define EPSC 1e-8f

// -------------------------------------------------------------------------
// Queue prep: f32 row-major [1050,128] -> bf16 WMMA B-matrix layout tiles.
// B order used here (contiguous K, A matches): lane = N + 16*(K%32>=16); VGPR v holds K pair
// (2v,2v+1) within the 16-K half; per lane per 32-K chunk = 32 contiguous B.
// Padded columns (>=1050) written as zero; masked by consumer.
// -------------------------------------------------------------------------
__global__ void pgc_queue_prep(const float* __restrict__ q,
                               char* __restrict__ bq) {
  int idx = blockIdx.x * blockDim.x + threadIdx.x;   // 1056*128 elements
  if (idx >= NCT * 16 * PROJ) return;
  int qi = idx >> 7;                  // padded column index 0..1055
  int k  = idx & 127;                 // K index
  float val = (qi < NQ) ? q[qi * PROJ + k] : 0.0f;
  int ct = qi >> 4, n = qi & 15;
  int c  = k >> 5,  kk = k & 31;      // 32-K chunk
  int hi = kk >> 4, pos = kk & 15;
  int lane = n + (hi << 4);
  int vg = pos >> 1, lo = pos & 1;
  size_t off = (size_t)ct * TILE_BYTES + (size_t)c * 1024 + lane * 32 + vg * 4 + lo * 2;
  *(__bf16*)(bq + off) = (__bf16)val;
}

// -------------------------------------------------------------------------
// Main kernel: one block (4 waves) per 16-sample tile.
// -------------------------------------------------------------------------
__global__ __launch_bounds__(128) void pgc_main(
    const float* __restrict__ act, const float* __restrict__ ema,
    const float* __restrict__ plab, const char* __restrict__ bq,
    float* __restrict__ blockSums) {
  __shared__ float actLDS[16][132];   // normalized activations (pad avoids conflicts)
  __shared__ float lpos_s[16];
  __shared__ int   lab_s[16];
  __shared__ float sDen[4][16];
  __shared__ float sPos[4][16];
  __shared__ float sPer[16];

  const int tid    = threadIdx.x;
  const int w      = tid >> 5;        // wave id 0..3
  const int lane   = tid & 31;
  const int laneHi = lane >> 4;
  const int base   = blockIdx.x * 16;

  // ---- Stage 1: normalize 16 rows, compute l_pos (wave w owns samples 4w..4w+3)
  for (int r = 0; r < 4; ++r) {
    int m = w * 4 + r;
    const float4* a4 = (const float4*)(act + (size_t)(base + m) * PROJ);
    const float4* e4 = (const float4*)(ema + (size_t)(base + m) * PROJ);
    float4 av = a4[lane];
    float4 ev = e4[lane];
    float sa = av.x*av.x + av.y*av.y + av.z*av.z + av.w*av.w;
    float se = ev.x*ev.x + ev.y*ev.y + ev.z*ev.z + ev.w*ev.w;
    float sd = av.x*ev.x + av.y*ev.y + av.z*ev.z + av.w*ev.w;
    for (int mk = 1; mk < 32; mk <<= 1) {
      sa += __shfl_xor(sa, mk, 32);
      se += __shfl_xor(se, mk, 32);
      sd += __shfl_xor(sd, mk, 32);
    }
    float inva = 1.0f / fmaxf(sqrtf(sa), 1e-12f);
    float inve = 1.0f / fmaxf(sqrtf(se), 1e-12f);
    float* row = actLDS[m];
    row[lane*4 + 0] = av.x * inva;
    row[lane*4 + 1] = av.y * inva;
    row[lane*4 + 2] = av.z * inva;
    row[lane*4 + 3] = av.w * inva;
    if (lane == 0) lpos_s[m] = __expf(2.0f * sd * inva * inve);  // T=0.5
  }
  if (tid < 16) {   // argmax over 7 logits (first-max wins, matches jnp.argmax)
    const float* p = plab + (size_t)(base + tid) * CLS;
    int best = 0; float bv = p[0];
    for (int j = 1; j < CLS; ++j) { float v = p[j]; if (v > bv) { bv = v; best = j; } }
    lab_s[tid] = best;
  }
  __syncthreads();

  // ---- Stage 2: build bf16 A fragments (16x128, 4 chunks of K=32)
  // A layout: lane holds M=lane&15; element e2 -> K = 16*laneHi + e2 (matches B).
  union AF { v16bf v; __bf16 e[16]; };
  AF afrag[4];
  const float* arow = actLDS[lane & 15];
  #pragma unroll
  for (int c = 0; c < 4; ++c) {
    #pragma unroll
    for (int e2 = 0; e2 < 16; ++e2) {
      int vg = e2 >> 1, lo = e2 & 1;
      int K = c * 32 + (laneHi << 4) + (vg << 1) + lo;
      afrag[c].e[e2] = (__bf16)arow[K];
    }
  }
  int labv[8];
  #pragma unroll
  for (int v = 0; v < 8; ++v) labv[v] = lab_s[(laneHi << 3) + v];

  // ---- Stage 3: GEMM over column tiles, fused exp/mask accumulation
  //      (fully branchless inner body: masks via cndmask + fma, no EXEC churn)
  float accD[8] = {0.f,0.f,0.f,0.f,0.f,0.f,0.f,0.f};
  float accP[8] = {0.f,0.f,0.f,0.f,0.f,0.f,0.f,0.f};
  const int n = lane & 15;
  for (int ct = w; ct < NCT; ct += 4) {
    int pct = (ct + 4 < NCT) ? (ct + 4) : ct;     // clamped -> unconditional prefetch
    __builtin_prefetch(bq + (size_t)pct * TILE_BYTES + lane * 32, 0, 1);
    v8f acc = {0.f,0.f,0.f,0.f,0.f,0.f,0.f,0.f};
    #pragma unroll
    for (int c = 0; c < 4; ++c) {
      v16bf b = *(const v16bf*)(bq + (size_t)ct * TILE_BYTES + (size_t)c * 1024 + lane * 32);
      acc = __builtin_amdgcn_wmma_f32_16x16x32_bf16(
          false, afrag[c].v, false, b, (short)0, acc, false, false);
    }
    union CF { v8f v; float f[8]; } cf; cf.v = acc;
    int q = ct * 16 + n;                          // queue row owned by this lane
    float validf = (q < NQ) ? 1.0f : 0.0f;        // pad-column mask (last tile only)
    int cl = q / QS;                              // class of this queue row
    #pragma unroll
    for (int v = 0; v < 8; ++v) {
      float s = cf.f[v];        // C layout: VGPR v -> sample M = v + 8*laneHi
      float e = __expf(2.0f * s);                 // sim = exp(s/T)
      accD[v] = fmaf(validf, e, accD[v]);
      float pm = (cl == labv[v]) ? validf : 0.0f; // positive-class mask
      accP[v] = fmaf(pm, s, accP[v]);             // raw-dot sum over positive rows
    }
  }

  // ---- Stage 4: reduce over the 16 lanes (N) of each half -> per-sample sums
  #pragma unroll
  for (int v = 0; v < 8; ++v) {
    float d = accD[v], p = accP[v];
    for (int mk = 1; mk < 16; mk <<= 1) {
      d += __shfl_xor(d, mk, 32);
      p += __shfl_xor(p, mk, 32);
    }
    accD[v] = d; accP[v] = p;
  }
  if ((lane & 15) == 0) {
    int mo = laneHi << 3;
    #pragma unroll
    for (int v = 0; v < 8; ++v) { sDen[w][mo + v] = accD[v]; sPos[w][mo + v] = accP[v]; }
  }
  __syncthreads();

  // ---- Stage 5: per-sample loss (fixed summation order -> deterministic)
  if (tid < 16) {
    float ds = sDen[0][tid] + sDen[1][tid] + sDen[2][tid] + sDen[3][tid];
    float ps = sPos[0][tid] + sPos[1][tid] + sPos[2][tid] + sPos[3][tid];
    float lp = lpos_s[tid];
    float denom = lp + ds;
    float t0 = -__logf(lp / denom + EPSC);
    // sum over 150 positive rows of -log(sim/denom+eps) ~= 150*log(denom) - 2*sum(s)
    float per = (t0 + (float)QS * __logf(denom) - 2.0f * ps) / (float)(QS + 1);
    sPer[tid] = per;
  }
  __syncthreads();
  if (tid == 0) {
    float s = 0.f;
    for (int m = 0; m < 16; ++m) s += sPer[m];
    blockSums[blockIdx.x] = s;
  }
}

// -------------------------------------------------------------------------
// Final deterministic reduction -> mean
// -------------------------------------------------------------------------
__global__ __launch_bounds__(256) void pgc_final(const float* __restrict__ bs,
                                                 int nblk, int B,
                                                 float* __restrict__ out) {
  __shared__ float red[256];
  int tid = threadIdx.x;
  float s = 0.f;
  for (int i = tid; i < nblk; i += 256) s += bs[i];
  red[tid] = s;
  __syncthreads();
  for (int off = 128; off > 0; off >>= 1) {
    if (tid < off) red[tid] += red[tid + off];
    __syncthreads();
  }
  if (tid == 0) out[0] = red[0] / (float)B;
}

extern "C" void kernel_launch(void* const* d_in, const int* in_sizes, int n_in,
                              void* d_out, int out_size, void* d_ws, size_t ws_size,
                              hipStream_t stream) {
  const float* act   = (const float*)d_in[0];
  const float* ema   = (const float*)d_in[1];
  const float* plab  = (const float*)d_in[2];
  const float* qlist = (const float*)d_in[3];

  int B = in_sizes[0] / PROJ;          // 65536
  int nblk = B / 16;                   // 4096 sample tiles

  char*  wsB       = (char*)d_ws;      // bf16 B-layout queue tiles: 66*4096 B
  float* blockSums = (float*)(wsB + 512 * 1024);

  int nelem = NCT * 16 * PROJ;
  pgc_queue_prep<<<(nelem + 255) / 256, 256, 0, stream>>>(qlist, wsB);
  pgc_main<<<nblk, 128, 0, stream>>>(act, ema, plab, wsB, blockSums);
  pgc_final<<<1, 256, 0, stream>>>(blockSums, nblk, B, (float*)d_out);
}
